// GraphAttentionNetwork_37271726195317
// MI455X (gfx1250) — matrix operand order repaired and checked by reading, not compiled, not run on hardware
//
#include <hip/hip_runtime.h>

#define NN    20000
#define EE    320000
#define ETOT  (EE + NN)      // edges + self loops
#define INC   256
#define HIDC  64
#define HEADS 4
#define HC    (HEADS * HIDC) // 256
#define OUTC  64
#define SLOPE 0.2f

typedef __attribute__((ext_vector_type(16))) _Float16 v16h;
typedef __attribute__((ext_vector_type(8)))  float    v8f;

// ---------- order-preserving float <-> uint for atomic max ----------
__device__ __forceinline__ unsigned f2ord(float f) {
  unsigned u = __float_as_uint(f);
  return (u & 0x80000000u) ? ~u : (u | 0x80000000u);
}
__device__ __forceinline__ float ord2f(unsigned u) {
  unsigned v = (u & 0x80000000u) ? (u ^ 0x80000000u) : ~u;
  return __uint_as_float(v);
}

// ---------- small utility kernels ----------
__global__ void fill_f32(float* p, float v, int n) {
  int t = blockIdx.x * blockDim.x + threadIdx.x;
  if (t < n) p[t] = v;
}
__global__ void init_out_bias(float* out, const float* __restrict__ b, int n, int C) {
  int t = blockIdx.x * blockDim.x + threadIdx.x;
  if (t < n) out[t] = b[t % C];
}

// ---------- WMMA GEMM: C[M,Nc] = A[M,K] @ B[K,Nc] (+bias per column) ----------
// one wave32 per 16x16 tile, f32 data converted to f16 on the fly, f32 accum.
__global__ __launch_bounds__(256) void wmma_gemm_f16(
    const float* __restrict__ A, const float* __restrict__ B,
    const float* __restrict__ bias, float* __restrict__ C,
    int M, int Nc, int K) {
  int gwave = (blockIdx.x * blockDim.x + threadIdx.x) >> 5;
  int lane  = threadIdx.x & 31;
  int tilesN = Nc >> 4;
  int tm = gwave / tilesN;
  int tn = gwave % tilesN;
  if (tm * 16 >= M) return;                 // wave-uniform: EXEC stays all-1s
  int half = lane >> 4;
  int l15  = lane & 15;
  int mrow = tm * 16 + l15;
  const float* arow = A + (size_t)mrow * K;
  v8f acc = {};
  for (int k0 = 0; k0 < K; k0 += 32) {
    v16h a, b;
    // A: 16-bit A-matrix 16x32 layout (ISA 7.12.2)
#pragma unroll
    for (int j = 0; j < 8; ++j) {
      int kb = k0 + ((j < 4) ? 0 : 8) + 2 * j + 8 * half;
      a[2 * j]     = (_Float16)arow[kb];
      a[2 * j + 1] = (_Float16)arow[kb + 1];
    }
    // B: lane holds column N = tn*16+l15, K rows in the same order as A
    const float* brow = B + (size_t)k0 * Nc + tn * 16 + l15;
#pragma unroll
    for (int e = 0; e < 16; ++e) b[e] = (_Float16)brow[(size_t)(((e < 8) ? e : e + 8) + 8 * half) * Nc];
    acc = __builtin_amdgcn_wmma_f32_16x16x32_f16(
        false, a, false, b, (short)0, acc, false, false);
  }
  int n = tn * 16 + l15;
  float bv = bias ? bias[n] : 0.0f;
#pragma unroll
  for (int r = 0; r < 8; ++r) {
    int m = tm * 16 + r + 8 * half;
    C[(size_t)m * Nc + n] = acc[r] + bv;
  }
}

// ---------- per-(node,head) attention scores ----------
__global__ void scores_kernel(const float* __restrict__ h,
                              const float* __restrict__ asrc,
                              const float* __restrict__ adst,
                              float* __restrict__ es, float* __restrict__ ed,
                              int nodes, int heads, int C) {
  int t = blockIdx.x * blockDim.x + threadIdx.x;
  if (t >= nodes * heads) return;
  int n = t / heads, hh = t % heads;
  const float* hp = h + (size_t)n * heads * C + (size_t)hh * C;
  const float* ap = asrc + hh * C;
  const float* bp = adst + hh * C;
  float s0 = 0.f, s1 = 0.f;
  for (int c = 0; c < C; ++c) { s0 += hp[c] * ap[c]; s1 += hp[c] * bp[c]; }
  es[t] = s0; ed[t] = s1;
}

// ---------- edge pass 1: leaky-relu logits + segment max (atomic, ordered-uint) ----
__global__ void edge_logits_max(const int* __restrict__ ei,
                                const float* __restrict__ es,
                                const float* __restrict__ ed,
                                float* __restrict__ lg, unsigned* __restrict__ m,
                                int heads) {
  int t = blockIdx.x * blockDim.x + threadIdx.x;
  if (t >= ETOT * heads) return;
  int e = t / heads, hh = t % heads;
  int s = (e < EE) ? ei[e] : (e - EE);
  int d = (e < EE) ? ei[EE + e] : (e - EE);
  float l = es[s * heads + hh] + ed[d * heads + hh];
  l = (l > 0.f) ? l : SLOPE * l;
  lg[t] = l;
  atomicMax(&m[d * heads + hh], f2ord(l));
}

// ---------- edge pass 2: p = exp(l - m[dst]); z[dst] += p ----------
__global__ void edge_exp_sum(const int* __restrict__ ei,
                             const unsigned* __restrict__ m,
                             float* __restrict__ lg, float* __restrict__ z,
                             int heads) {
  int t = blockIdx.x * blockDim.x + threadIdx.x;
  if (t >= ETOT * heads) return;
  int e = t / heads, hh = t % heads;
  int d = (e < EE) ? ei[EE + e] : (e - EE);
  float p = expf(lg[t] - ord2f(m[d * heads + hh]));
  lg[t] = p;
  atomicAdd(&z[d * heads + hh], p);
}

// ---------- edge pass 3 (vectorized): agg[dst,ch..ch+3] += h[src,ch..ch+3]*alpha ----
// One thread per float4 of channels: edge indices + alpha amortized 4x,
// h[src] fetched with a single b128 load. Channel quads never straddle heads
// (Cph is a multiple of 4).
__global__ void edge_aggregate4(const int* __restrict__ ei,
                                const float* __restrict__ lg,
                                const float* __restrict__ z,
                                const float* __restrict__ h,
                                float* __restrict__ agg,
                                int heads, int Cph) {
  int Ctot = heads * Cph;
  int C4   = Ctot >> 2;
  long long t = (long long)blockIdx.x * blockDim.x + threadIdx.x;
  if (t >= (long long)ETOT * C4) return;
  int e  = (int)(t / C4);
  int c4 = (int)(t % C4);
  int ch = c4 << 2;
  int hh = ch / Cph;
  int s = (e < EE) ? ei[e] : (e - EE);
  int d = (e < EE) ? ei[EE + e] : (e - EE);
  float alpha = lg[e * heads + hh] / z[d * heads + hh];
  const float4 hv = *(const float4*)(h + (size_t)s * Ctot + ch);
  float* ap = agg + (size_t)d * Ctot + ch;
  atomicAdd(ap + 0, hv.x * alpha);
  atomicAdd(ap + 1, hv.y * alpha);
  atomicAdd(ap + 2, hv.z * alpha);
  atomicAdd(ap + 3, hv.w * alpha);
}

// ---------- bias + ELU (in place) ----------
__global__ void bias_elu(float* a, const float* __restrict__ b, int n, int C) {
  int t = blockIdx.x * blockDim.x + threadIdx.x;
  if (t < n) {
    float v = a[t] + b[t % C];
    a[t] = (v > 0.f) ? v : (expf(v) - 1.f);
  }
}

static inline unsigned nblk(long long n, int bt) { return (unsigned)((n + bt - 1) / bt); }

extern "C" void kernel_launch(void* const* d_in, const int* in_sizes, int n_in,
                              void* d_out, int out_size, void* d_ws, size_t ws_size,
                              hipStream_t stream) {
  const float* x   = (const float*)d_in[0];
  const int*   ei  = (const int*)d_in[1];   // [2, E] flat: src row then dst row
  const float* W1  = (const float*)d_in[2];
  const float* a1s = (const float*)d_in[3];
  const float* a1d = (const float*)d_in[4];
  const float* b1  = (const float*)d_in[5];
  const float* W2  = (const float*)d_in[6];
  const float* a2s = (const float*)d_in[7];
  const float* a2d = (const float*)d_in[8];
  const float* b2  = (const float*)d_in[9];
  float* out = (float*)d_out;

  float* ws = (float*)d_ws;
  size_t o = 0;
  float* h1   = ws + o; o += (size_t)NN * HC;     // x @ W1
  float* agg  = ws + o; o += (size_t)NN * HC;     // layer1 aggregate -> elu act
  float* h2   = ws + o; o += (size_t)NN * OUTC;   // act @ W2
  float* e1s  = ws + o; o += NN * HEADS;
  float* e1d  = ws + o; o += NN * HEADS;
  float* m1   = ws + o; o += NN * HEADS;          // ordered-uint max
  float* z1   = ws + o; o += NN * HEADS;
  float* lg1  = ws + o; o += (size_t)ETOT * HEADS;
  float* e2s  = ws + o; o += NN;
  float* e2d  = ws + o; o += NN;
  float* m2   = ws + o; o += NN;
  float* z2   = ws + o; o += NN;
  float* lg2  = ws + o; o += ETOT;
  (void)ws_size; (void)in_sizes; (void)n_in; (void)out_size;

  const int BT = 256;

  // ---- Layer 1 ----
  {
    long long waves = (long long)(NN / 16) * (HC / 16);   // 20000 tiles
    wmma_gemm_f16<<<nblk(waves * 32, BT), BT, 0, stream>>>(x, W1, nullptr, h1, NN, HC, INC);
  }
  scores_kernel<<<nblk((long long)NN * HEADS, BT), BT, 0, stream>>>(h1, a1s, a1d, e1s, e1d, NN, HEADS, HIDC);
  fill_f32<<<nblk(NN * HEADS, BT), BT, 0, stream>>>(m1, 0.0f, NN * HEADS); // bits=0 == below any finite encode
  fill_f32<<<nblk(NN * HEADS, BT), BT, 0, stream>>>(z1, 0.0f, NN * HEADS);
  fill_f32<<<nblk((long long)NN * HC, BT), BT, 0, stream>>>(agg, 0.0f, NN * HC);
  edge_logits_max<<<nblk((long long)ETOT * HEADS, BT), BT, 0, stream>>>(ei, e1s, e1d, lg1, (unsigned*)m1, HEADS);
  edge_exp_sum<<<nblk((long long)ETOT * HEADS, BT), BT, 0, stream>>>(ei, (const unsigned*)m1, lg1, z1, HEADS);
  edge_aggregate4<<<nblk((long long)ETOT * (HC / 4), BT), BT, 0, stream>>>(ei, lg1, z1, h1, agg, HEADS, HIDC);
  bias_elu<<<nblk((long long)NN * HC, BT), BT, 0, stream>>>(agg, b1, NN * HC, HC);

  // ---- Layer 2 ----
  {
    long long waves = (long long)(NN / 16) * (OUTC / 16); // 5000 tiles
    wmma_gemm_f16<<<nblk(waves * 32, BT), BT, 0, stream>>>(agg, W2, nullptr, h2, NN, OUTC, HC);
  }
  scores_kernel<<<nblk(NN, BT), BT, 0, stream>>>(h2, a2s, a2d, e2s, e2d, NN, 1, OUTC);
  fill_f32<<<nblk(NN, BT), BT, 0, stream>>>(m2, 0.0f, NN);
  fill_f32<<<nblk(NN, BT), BT, 0, stream>>>(z2, 0.0f, NN);
  init_out_bias<<<nblk((long long)NN * OUTC, BT), BT, 0, stream>>>(out, b2, NN * OUTC, OUTC);
  edge_logits_max<<<nblk(ETOT, BT), BT, 0, stream>>>(ei, e2s, e2d, lg2, (unsigned*)m2, 1);
  edge_exp_sum<<<nblk(ETOT, BT), BT, 0, stream>>>(ei, (const unsigned*)m2, lg2, z2, 1);
  edge_aggregate4<<<nblk((long long)ETOT * (OUTC / 4), BT), BT, 0, stream>>>(ei, lg2, z2, h2, out, 1, OUTC);
}
